// PershomLearnedFilt_43482248905287
// MI455X (gfx1250) — compile-verified, matrix operand order checked by reading, not run on hardware
//
#include <hip/hip_runtime.h>

// ---------------- problem constants (match reference) ----------------
#define NN      102400          // total nodes
#define GG      256             // graphs
#define NPG     400             // nodes per graph
#define EE      (NN * 16)       // directed edges
#define EPG     (EE / GG)       // 6400 edges per graph
#define PPG     399             // finite H0 pairs per graph
#define E1PG    50              // essential H1 points per graph
#define SCENT   128             // struct elements
#define SLOPE   0.01f
#define BN_EPS  1e-5f

typedef _Float16 v8h  __attribute__((ext_vector_type(8)));
typedef _Float16 v16h __attribute__((ext_vector_type(16)));
typedef float    v8f  __attribute__((ext_vector_type(8)));

union V16 { v16h v; v8h h[2]; };

__device__ __forceinline__ float lrelu(float x) { return x >= 0.f ? x : SLOPE * x; }

// ---------------- weight prep: f32 [K,Kout] -> f16 transposed [Kout,K] ----------------
__global__ void conv_wt(const float* __restrict__ W, _Float16* __restrict__ Wt, int K, int Kout) {
    int i = blockIdx.x * blockDim.x + threadIdx.x;
    if (i < K * Kout) {
        int k = i / Kout, n = i % Kout;
        Wt[(size_t)n * K + k] = (_Float16)W[i];
    }
}

__global__ void zero_kernel(float* p, int n) {
    int i = blockIdx.x * blockDim.x + threadIdx.x;
    if (i < n) p[i] = 0.f;
}

// ---------------- embedding: z0 = concat(embed_deg[deg], embed_lab[lab]) -> xh[:,0:128] f16 ----
__global__ void embed_kernel(const int* __restrict__ deg, const int* __restrict__ lab,
                             const float* __restrict__ wd, const float* __restrict__ wl,
                             _Float16* __restrict__ xh) {
    int node = blockIdx.x;         // NN blocks
    int c = threadIdx.x;           // 128 threads
    float v = (c < 64) ? wd[deg[node] * 64 + c] : wl[lab[node] * 64 + (c - 64)];
    xh[(size_t)node * 256 + c] = (_Float16)v;
}

// ---------------- per-graph edge aggregation into LDS, out = (1+eps)*x + agg  (f16) ---------
// One workgroup per graph. LDS accumulator: NPG*D floats (204.8 KB for D=128, fits 320KB WGP LDS).
// Sources are graph-local and the whole activation tensor is L2-resident -> gathers hit L2;
// scatter is ds_add_f32 (LDS atomics), no global atomic traffic.
template <int D>
__global__ void agg_kernel(const _Float16* __restrict__ xh, int colOff,
                           const float* __restrict__ epsPtr,
                           const int* __restrict__ esrc, const int* __restrict__ edst,
                           _Float16* __restrict__ outA) {
    extern __shared__ float lds[];                 // NPG * D
    const int g = blockIdx.x, tid = threadIdx.x, nt = blockDim.x;
    for (int i = tid; i < NPG * D; i += nt) lds[i] = 0.f;
    __syncthreads();
    const int base = g * EPG;
    const int CH = D / 4;                          // 4-float chunks per edge vector
    for (int w = tid; w < EPG * CH; w += nt) {
        int e = w / CH, ch = w % CH;
        int s = esrc[base + e], d = edst[base + e];
        const _Float16* sp = xh + (size_t)s * 256 + colOff + ch * 4;
        float* dp = lds + (size_t)(d - g * NPG) * D + ch * 4;
        atomicAdd(&dp[0], (float)sp[0]);
        atomicAdd(&dp[1], (float)sp[1]);
        atomicAdd(&dp[2], (float)sp[2]);
        atomicAdd(&dp[3], (float)sp[3]);
    }
    __syncthreads();
    const float ope = 1.f + *epsPtr;
    for (int i = tid; i < NPG * D; i += nt) {
        int nloc = i / D, c = i % D;
        size_t node = (size_t)g * NPG + nloc;
        float x = (float)xh[node * 256 + colOff + c];
        outA[node * (size_t)D + c] = (_Float16)(ope * x + lds[i]);
    }
}

// ---------------- WMMA GEMM: one wave computes a 16x16 tile of  act(A[N,K] @ W[K,Kout] + b) ----
// A row-major f16 (row stride == K); Bt = W^T row-major f16 [Kout,K] so both fragments load as
// two contiguous 16B chunks per lane matching the 16-bit A/B VGPR striping:
//   lanes 0-15 : K = kb+0..7   and kb+16..23
//   lanes 16-31: K = kb+8..15  and kb+24..31
__global__ __launch_bounds__(32) void wmma_gemm(
    const _Float16* __restrict__ A, const _Float16* __restrict__ Bt,
    const float* __restrict__ bias, int K,
    float* __restrict__ outF, int strideF,
    _Float16* __restrict__ outH, int strideH, int colOff, int act) {
    const int tm = blockIdx.x, tn = blockIdx.y;
    const int lane = threadIdx.x, rlo = lane & 15, sel = lane >> 4;
    const _Float16* ar = A + (size_t)(tm * 16 + rlo) * K;
    const _Float16* br = Bt + (size_t)(tn * 16 + rlo) * K;
    v8f acc = {};
    for (int kb = 0; kb < K; kb += 32) {
        __builtin_prefetch(ar + kb + 64, 0, 1);
        V16 a, b;
        a.h[0] = *(const v8h*)(ar + kb + sel * 8);
        a.h[1] = *(const v8h*)(ar + kb + 16 + sel * 8);
        b.h[0] = *(const v8h*)(br + kb + sel * 8);
        b.h[1] = *(const v8h*)(br + kb + 16 + sel * 8);
        acc = __builtin_amdgcn_wmma_f32_16x16x32_f16(
            false, a.v, false, b.v, (short)0, acc, false, false);
    }
    const int col = tn * 16 + rlo;
    const float bv = bias ? bias[col] : 0.f;
#pragma unroll
    for (int i = 0; i < 8; i++) {
        int m = tm * 16 + i + sel * 8;           // C/D layout: vgpr i -> row i (+8 for hi lanes)
        float v = acc[i] + bv;
        if (act) v = lrelu(v);
        if (outF) outF[(size_t)m * strideF + col] = v;
        if (outH) outH[(size_t)m * strideH + colOff + col] = (_Float16)v;
    }
}

// ---------------- per-channel sum / sumsq over N for BatchNorm (train-mode stats) -------------
__global__ void stats_kernel(const float* __restrict__ h, float* __restrict__ out /*[128]*/) {
    __shared__ float s1[256], s2[256];
    int t = threadIdx.x, col = t & 63, grp = t >> 6;     // 4 row groups x 64 cols
    float s = 0.f, ss = 0.f;
    for (int r = blockIdx.x * 4 + grp; r < NN; r += gridDim.x * 4) {
        float v = h[(size_t)r * 64 + col];
        s += v; ss += v * v;
    }
    s1[t] = s; s2[t] = ss;
    __syncthreads();
    if (t < 64) {
        float a = s1[t] + s1[t + 64] + s1[t + 128] + s1[t + 192];
        float b = s2[t] + s2[t + 64] + s2[t + 128] + s2[t + 192];
        atomicAdd(&out[col], a);
        atomicAdd(&out[64 + col], b);
    }
}

// ---------------- BN finalize + LeakyReLU -> f16 at (outH + node*stride + colOff) -------------
__global__ void bn_fin(const float* __restrict__ h, const float* __restrict__ st,
                       const float* __restrict__ g, const float* __restrict__ b,
                       _Float16* __restrict__ outH, int stride, int colOff) {
    int i = blockIdx.x * blockDim.x + threadIdx.x;
    if (i >= NN * 64) return;
    int node = i >> 6, c = i & 63;
    float m = st[c] * (1.f / NN);
    float var = st[64 + c] * (1.f / NN) - m * m;         // biased var (matches jnp.var)
    float v = (h[i] - m) * rsqrtf(var + BN_EPS) * g[c] + b[c];
    outH[(size_t)node * stride + colOff + c] = (_Float16)lrelu(v);
}

// ---------------- node filtration: nf = sigmoid(fcH @ w2 + b2) --------------------------------
__global__ void nf_kernel(const _Float16* __restrict__ fcH, const float* __restrict__ w,
                          const float* __restrict__ b, float* __restrict__ nf) {
    int i = blockIdx.x * blockDim.x + threadIdx.x;
    if (i >= NN) return;
    float acc = b[0];
#pragma unroll
    for (int j = 0; j < 64; j++) acc += (float)fcH[(size_t)i * 64 + j] * w[j];
    nf[i] = 1.f / (1.f + expf(-acc));
}

// ---------------- SLayerRationalHat, finite H0 pairs (2-D points), block = graph --------------
__global__ void slayer_pairs(const float* __restrict__ nf, const int* __restrict__ pidx,
                             const float* __restrict__ c0, const float* __restrict__ r0,
                             float* __restrict__ feats) {
    __shared__ float p0s[PPG], p1s[PPG];
    int g = blockIdx.x, t = threadIdx.x;                 // 128 threads, t = struct element
    int base = g * PPG;
    for (int i = t; i < PPG; i += 128) {
        p0s[i] = nf[pidx[2 * (base + i) + 0]];
        p1s[i] = nf[pidx[2 * (base + i) + 1]];
    }
    __syncthreads();
    float cx = c0[2 * t], cy = c0[2 * t + 1];
    float r = fabsf(r0[0]);
    float acc = 0.f;
    for (int i = 0; i < PPG; i++) {
        float d = fabsf(p0s[i] - cx) + fabsf(p1s[i] - cy);
        acc += 1.f / (1.f + d) - 1.f / (1.f + fabsf(r - d));
    }
    feats[(size_t)g * (3 * SCENT) + t] = acc;
}

// ---------------- SLayer, essential (1-D points), pp points per graph -------------------------
__global__ void slayer_ess(const float* __restrict__ nf, const int* __restrict__ idx, int pp,
                           const float* __restrict__ ce, const float* __restrict__ re,
                           float* __restrict__ feats, int colOff) {
    __shared__ float ps[64];
    int g = blockIdx.x, t = threadIdx.x;                 // 128 threads
    for (int i = t; i < pp; i += 128) ps[i] = nf[idx[g * pp + i]];
    __syncthreads();
    float c = ce[t];
    float r = fabsf(re[0]);
    float acc = 0.f;
    for (int i = 0; i < pp; i++) {
        float d = fabsf(ps[i] - c);
        acc += 1.f / (1.f + d) - 1.f / (1.f + fabsf(r - d));
    }
    feats[(size_t)g * (3 * SCENT) + colOff + t] = acc;
}

// ---------------- classifier: lrelu(feats @ w1 + b1) @ w2 + b2 -> out[G,6] --------------------
__global__ void cls_kernel(const float* __restrict__ feats,
                           const float* __restrict__ w1, const float* __restrict__ b1,
                           const float* __restrict__ w2, const float* __restrict__ b2,
                           float* __restrict__ out) {
    __shared__ float hid[64];
    int g = blockIdx.x, t = threadIdx.x;                 // 64 threads
    const float* f = feats + (size_t)g * (3 * SCENT);
    float acc = b1[t];
    for (int k = 0; k < 3 * SCENT; k++) acc += f[k] * w1[k * 64 + t];
    hid[t] = lrelu(acc);
    __syncthreads();
    if (t < 6) {
        float o = b2[t];
#pragma unroll
        for (int j = 0; j < 64; j++) o += hid[j] * w2[j * 6 + t];
        out[(size_t)g * 6 + t] = o;
    }
}

// =============================================================================================
extern "C" void kernel_launch(void* const* d_in, const int* in_sizes, int n_in,
                              void* d_out, int out_size, void* d_ws, size_t ws_size,
                              hipStream_t stream) {
    (void)in_sizes; (void)n_in; (void)out_size; (void)ws_size;
    // ---- inputs (setup_inputs order) ----
    const int*   node_deg = (const int*)d_in[0];
    const int*   node_lab = (const int*)d_in[1];
    const int*   edge_src = (const int*)d_in[2];
    const int*   edge_dst = (const int*)d_in[3];
    const int*   pair_idx = (const int*)d_in[4];
    const int*   ess0_idx = (const int*)d_in[6];
    const int*   ess1_idx = (const int*)d_in[8];
    const float* emb_deg  = (const float*)d_in[10];
    const float* emb_lab  = (const float*)d_in[11];
    const float* eps0     = (const float*)d_in[12];
    const float* w1_0     = (const float*)d_in[13];
    const float* b1_0     = (const float*)d_in[14];
    const float* w2_0     = (const float*)d_in[15];
    const float* b2_0     = (const float*)d_in[16];
    const float* bn0_g    = (const float*)d_in[17];
    const float* bn0_b    = (const float*)d_in[18];
    const float* eps1     = (const float*)d_in[19];
    const float* w1_1     = (const float*)d_in[20];
    const float* b1_1     = (const float*)d_in[21];
    const float* w2_1     = (const float*)d_in[22];
    const float* b2_1     = (const float*)d_in[23];
    const float* bn1_g    = (const float*)d_in[24];
    const float* bn1_b    = (const float*)d_in[25];
    const float* fc_w1    = (const float*)d_in[26];
    const float* fc_b1    = (const float*)d_in[27];
    const float* fc_bn_g  = (const float*)d_in[28];
    const float* fc_bn_b  = (const float*)d_in[29];
    const float* fc_w2    = (const float*)d_in[30];
    const float* fc_b2    = (const float*)d_in[31];
    const float* c0       = (const float*)d_in[32];
    const float* r0       = (const float*)d_in[33];
    const float* ce0      = (const float*)d_in[34];
    const float* re0      = (const float*)d_in[35];
    const float* ce1      = (const float*)d_in[36];
    const float* re1      = (const float*)d_in[37];
    const float* cls_w1   = (const float*)d_in[38];
    const float* cls_b1   = (const float*)d_in[39];
    const float* cls_w2   = (const float*)d_in[40];
    const float* cls_b2   = (const float*)d_in[41];
    float* out = (float*)d_out;

    // ---- carve workspace (~145 MB) ----
    char* base = (char*)d_ws;
    size_t off = 0;
    auto take = [&](size_t bytes) -> void* {
        void* p = base + off;
        off = (off + bytes + 255) & ~(size_t)255;
        return p;
    };
    _Float16* xh    = (_Float16*)take((size_t)NN * 256 * 2); // concat buf: z0|z1|z2 f16 [N,256]
    _Float16* bufA  = (_Float16*)take((size_t)NN * 128 * 2); // GIN inputs (post-agg) f16
    _Float16* bufB  = (_Float16*)take((size_t)NN * 128 * 2); // MLP-mid f16
    float*    hraw  = (float*)  take((size_t)NN * 64 * 4);   // pre-BN f32
    _Float16* fcH   = (_Float16*)take((size_t)NN * 64 * 2);  // post fc-BN f16
    float*    nf    = (float*)  take((size_t)NN * 4);        // node filtration
    float*    feats = (float*)  take((size_t)GG * 3 * SCENT * 4);
    float*    stats = (float*)  take(3 * 128 * 4);           // {sum,sumsq} x 3 BN layers
    _Float16* wt0a  = (_Float16*)take(128 * 128 * 2);        // transposed f16 weights
    _Float16* wt0b  = (_Float16*)take(128 * 64 * 2);
    _Float16* wt1a  = (_Float16*)take(64 * 64 * 2);
    _Float16* wt1b  = (_Float16*)take(64 * 64 * 2);
    _Float16* wtfc  = (_Float16*)take(256 * 64 * 2);

    // ---- weight conversion + stats zero ----
    conv_wt<<<(128 * 128 + 255) / 256, 256, 0, stream>>>(w1_0, wt0a, 128, 128);
    conv_wt<<<(128 * 64 + 255) / 256, 256, 0, stream>>>(w2_0, wt0b, 128, 64);
    conv_wt<<<(64 * 64 + 255) / 256, 256, 0, stream>>>(w1_1, wt1a, 64, 64);
    conv_wt<<<(64 * 64 + 255) / 256, 256, 0, stream>>>(w2_1, wt1b, 64, 64);
    conv_wt<<<(256 * 64 + 255) / 256, 256, 0, stream>>>(fc_w1, wtfc, 256, 64);
    zero_kernel<<<2, 256, 0, stream>>>(stats, 3 * 128);

    // ---- z0 embedding into xh[:,0:128] ----
    embed_kernel<<<NN, 128, 0, stream>>>(node_deg, node_lab, emb_deg, emb_lab, xh);

    const dim3 wave(32, 1, 1);

    // ---- GIN layer 0 ----
    agg_kernel<128><<<GG, 1024, NPG * 128 * 4, stream>>>(xh, 0, eps0, edge_src, edge_dst, bufA);
    wmma_gemm<<<dim3(NN / 16, 8), wave, 0, stream>>>(bufA, wt0a, b1_0, 128,
                                                     nullptr, 0, bufB, 128, 0, 1);
    wmma_gemm<<<dim3(NN / 16, 4), wave, 0, stream>>>(bufB, wt0b, b2_0, 128,
                                                     hraw, 64, nullptr, 0, 0, 0);
    stats_kernel<<<1024, 256, 0, stream>>>(hraw, stats);
    bn_fin<<<(NN * 64 + 255) / 256, 256, 0, stream>>>(hraw, stats, bn0_g, bn0_b, xh, 256, 128);

    // ---- GIN layer 1 ----
    agg_kernel<64><<<GG, 1024, NPG * 64 * 4, stream>>>(xh, 128, eps1, edge_src, edge_dst, bufA);
    wmma_gemm<<<dim3(NN / 16, 4), wave, 0, stream>>>(bufA, wt1a, b1_1, 64,
                                                     nullptr, 0, bufB, 64, 0, 1);
    wmma_gemm<<<dim3(NN / 16, 4), wave, 0, stream>>>(bufB, wt1b, b2_1, 64,
                                                     hraw, 64, nullptr, 0, 0, 0);
    stats_kernel<<<1024, 256, 0, stream>>>(hraw, stats + 128);
    bn_fin<<<(NN * 64 + 255) / 256, 256, 0, stream>>>(hraw, stats + 128, bn1_g, bn1_b, xh, 256, 192);

    // ---- FC over concat [N,256] (xh holds z0|z1|z2 already) ----
    wmma_gemm<<<dim3(NN / 16, 4), wave, 0, stream>>>(xh, wtfc, fc_b1, 256,
                                                     hraw, 64, nullptr, 0, 0, 0);
    stats_kernel<<<1024, 256, 0, stream>>>(hraw, stats + 256);
    bn_fin<<<(NN * 64 + 255) / 256, 256, 0, stream>>>(hraw, stats + 256, fc_bn_g, fc_bn_b, fcH, 64, 0);
    nf_kernel<<<(NN + 255) / 256, 256, 0, stream>>>(fcH, fc_w2, fc_b2, nf);

    // ---- persistence features + classifier ----
    slayer_pairs<<<GG, 128, 0, stream>>>(nf, pair_idx, c0, r0, feats);
    slayer_ess<<<GG, 128, 0, stream>>>(nf, ess0_idx, 1, ce0, re0, feats, 128);
    slayer_ess<<<GG, 128, 0, stream>>>(nf, ess1_idx, E1PG, ce1, re1, feats, 256);
    cls_kernel<<<GG, 64, 0, stream>>>(feats, cls_w1, cls_b1, cls_w2, cls_b2, out);
}
